// MockDeepSeekAttention_33483565040120
// MI455X (gfx1250) — matrix operand order repaired and checked by reading, not compiled, not run on hardware
//
#include <hip/hip_runtime.h>
#include <cstdint>

// ---------------------------------------------------------------------------
// MockDeepSeekAttention (MLA) forward for gfx1250 / MI455X.
// All matmuls run on v_wmma_f32_16x16x32_bf16 (wave32 WMMA).
// GEMMs use double-buffered LDS (1 barrier / K-step) + global_prefetch.
//
// Shapes: B=1, S=2048, HIDDEN=4096, H=32, Q_HEAD=192 (NOPE 128 + ROPE 64),
//         KV_LORA=512, D_V=128.
// ---------------------------------------------------------------------------

#define S_LEN   2048
#define HIDDEN  4096
#define NHEAD   32
#define QHEAD   192
#define DNOPE   128
#define DROPE   64
#define DV      128
#define KVLORA  512
#define HQ      (NHEAD * QHEAD)          // 6144
#define CKV_N   (KVLORA + DROPE)         // 576
#define KV_N    (NHEAD * (DNOPE + DV))   // 8192
#define CTX_N   (NHEAD * DV)             // 4096
#define ATT_SCALE 0.0721687836487032f    // 1/sqrt(192)

typedef __bf16 bf16_t;
typedef __attribute__((ext_vector_type(16))) __bf16 v16bf;
typedef __attribute__((ext_vector_type(8)))  float  v8f;

union AFrag { v16bf v; uint4 q[2]; };

__device__ __forceinline__ uint16_t f2bfbits(float f) {
    uint32_t u = __builtin_bit_cast(uint32_t, f);
    u += 0x7fffu + ((u >> 16) & 1u);            // round-to-nearest-even
    return (uint16_t)(u >> 16);
}
__device__ __forceinline__ uint32_t pack2bf(float a, float b) {
    return (uint32_t)f2bfbits(a) | ((uint32_t)f2bfbits(b) << 16);
}
__device__ __forceinline__ v8f wmma_bf16(const AFrag& a, const AFrag& b, v8f c) {
    return __builtin_amdgcn_wmma_f32_16x16x32_bf16(false, a.v, false, b.v,
                                                   (short)0, c, false, false);
}

// ---------------------------------------------------------------------------
// Generic NT GEMM: C[M,N] = A[M,K] * B[N,K]^T.
// TA/TB selected by size: 4 -> f32 (converted to bf16 while staging to LDS),
// 2 -> bf16 (staged directly). OUT_BF16 selects C dtype.
// 256 threads, 128x128 tile, K-step 32, 8 waves in a 4(M) x 2(N) grid.
// Double-buffered LDS: global fetch of tile t+1 overlaps WMMAs of tile t;
// one barrier per K-step.
// ---------------------------------------------------------------------------
template <typename TA, typename TB, bool OUT_BF16>
__global__ __launch_bounds__(256) void gemm_nt(const void* __restrict__ Ap,
                                               const void* __restrict__ Bp,
                                               void* __restrict__ Cp,
                                               int M, int N, int K,
                                               int lda, int ldb, int ldc) {
    __shared__ uint16_t As[2][128][40];  // ping-pong; rows 80B (16B aligned)
    __shared__ uint16_t Bs[2][128][40];

    const int tid  = threadIdx.x;
    const int lane = tid & 31, wave = tid >> 5;
    const int hf   = lane >> 4, l16 = lane & 15;
    const int wm   = wave >> 1, wn  = wave & 1;
    const int m0   = blockIdx.y * 128, n0 = blockIdx.x * 128;

    const v8f vz = {0.f, 0.f, 0.f, 0.f, 0.f, 0.f, 0.f, 0.f};
    v8f acc[2][4];
    for (int i = 0; i < 2; ++i)
        for (int j = 0; j < 4; ++j) acc[i][j] = vz;

    // staging registers for the incoming tile
    float4 aF[4], bF[4];
    uint4  aH[2], bH[2];

    auto fetch = [&](int k0) {
        const bool pf = (k0 + 96 <= K);          // prefetch two K-steps ahead
        if constexpr (sizeof(TA) == 4) {
            const float* A = (const float*)Ap;
            #pragma unroll
            for (int i = 0; i < 4; ++i) {
                int c = tid + i * 256, row = c >> 3, col = (c & 7) * 4;
                int m = m0 + row;
                aF[i] = make_float4(0.f, 0.f, 0.f, 0.f);
                if (m < M) {
                    const float* p = A + (size_t)m * lda + k0 + col;
                    aF[i] = *(const float4*)p;
                    if (i == 0 && pf) __builtin_prefetch(p + 64, 0, 1);
                }
            }
        } else {
            const uint16_t* A = (const uint16_t*)Ap;
            #pragma unroll
            for (int i = 0; i < 2; ++i) {
                int c = tid + i * 256, row = c >> 2, col = (c & 3) * 8;
                int m = m0 + row;
                aH[i] = make_uint4(0u, 0u, 0u, 0u);
                if (m < M) {
                    const uint16_t* p = A + (size_t)m * lda + k0 + col;
                    aH[i] = *(const uint4*)p;
                    if (i == 0 && pf) __builtin_prefetch(p + 64, 0, 1);
                }
            }
        }
        if constexpr (sizeof(TB) == 4) {
            const float* B = (const float*)Bp;
            #pragma unroll
            for (int i = 0; i < 4; ++i) {
                int c = tid + i * 256, row = c >> 3, col = (c & 7) * 4;
                int n = n0 + row;
                bF[i] = make_float4(0.f, 0.f, 0.f, 0.f);
                if (n < N) {
                    const float* p = B + (size_t)n * ldb + k0 + col;
                    bF[i] = *(const float4*)p;
                    if (i == 0 && pf) __builtin_prefetch(p + 64, 0, 1);
                }
            }
        } else {
            const uint16_t* B = (const uint16_t*)Bp;
            #pragma unroll
            for (int i = 0; i < 2; ++i) {
                int c = tid + i * 256, row = c >> 2, col = (c & 3) * 8;
                int n = n0 + row;
                bH[i] = make_uint4(0u, 0u, 0u, 0u);
                if (n < N) {
                    const uint16_t* p = B + (size_t)n * ldb + k0 + col;
                    bH[i] = *(const uint4*)p;
                    if (i == 0 && pf) __builtin_prefetch(p + 64, 0, 1);
                }
            }
        }
    };

    auto stage = [&](int buf) {
        if constexpr (sizeof(TA) == 4) {
            #pragma unroll
            for (int i = 0; i < 4; ++i) {
                int c = tid + i * 256, row = c >> 3, col = (c & 7) * 4;
                uint2 p; p.x = pack2bf(aF[i].x, aF[i].y); p.y = pack2bf(aF[i].z, aF[i].w);
                *(uint2*)&As[buf][row][col] = p;
            }
        } else {
            #pragma unroll
            for (int i = 0; i < 2; ++i) {
                int c = tid + i * 256, row = c >> 2, col = (c & 3) * 8;
                *(uint4*)&As[buf][row][col] = aH[i];
            }
        }
        if constexpr (sizeof(TB) == 4) {
            #pragma unroll
            for (int i = 0; i < 4; ++i) {
                int c = tid + i * 256, row = c >> 3, col = (c & 7) * 4;
                uint2 p; p.x = pack2bf(bF[i].x, bF[i].y); p.y = pack2bf(bF[i].z, bF[i].w);
                *(uint2*)&Bs[buf][row][col] = p;
            }
        } else {
            #pragma unroll
            for (int i = 0; i < 2; ++i) {
                int c = tid + i * 256, row = c >> 2, col = (c & 3) * 8;
                *(uint4*)&Bs[buf][row][col] = bH[i];
            }
        }
    };

    const int T = K >> 5;                 // K-steps of 32
    fetch(0);
    stage(0);
    __syncthreads();

    for (int t = 0; t < T; ++t) {
        if (t + 1 < T) fetch((t + 1) * 32);   // overlap with WMMAs below

        const int cur = t & 1;
        AFrag a[2], b[4];
        #pragma unroll
        for (int mi = 0; mi < 2; ++mi) {
            int r = wm * 32 + mi * 16 + l16;
            a[mi].q[0] = *(const uint4*)&As[cur][r][8 * hf];      // k = 8h..8h+7
            a[mi].q[1] = *(const uint4*)&As[cur][r][16 + 8 * hf]; // k = 16+8h..
        }
        #pragma unroll
        for (int ni = 0; ni < 4; ++ni) {
            int r = wn * 64 + ni * 16 + l16;
            b[ni].q[0] = *(const uint4*)&Bs[cur][r][8 * hf];      // k = 8h..8h+7
            b[ni].q[1] = *(const uint4*)&Bs[cur][r][16 + 8 * hf];
        }
        #pragma unroll
        for (int mi = 0; mi < 2; ++mi)
            #pragma unroll
            for (int ni = 0; ni < 4; ++ni)
                acc[mi][ni] = wmma_bf16(a[mi], b[ni], acc[mi][ni]);

        if (t + 1 < T) stage((t + 1) & 1);    // writes opposite buffer
        __syncthreads();
    }

    // ---- store C (f32 C/D layout: m = r + 8*hf, n = l16) ----
    #pragma unroll
    for (int mi = 0; mi < 2; ++mi)
        #pragma unroll
        for (int ni = 0; ni < 4; ++ni)
            #pragma unroll
            for (int r = 0; r < 8; ++r) {
                int m = m0 + wm * 32 + mi * 16 + r + 8 * hf;
                int n = n0 + wn * 64 + ni * 16 + l16;
                if (m < M && n < N) {
                    float val = acc[mi][ni][r];
                    if constexpr (OUT_BF16)
                        ((uint16_t*)Cp)[(size_t)m * ldc + n] = f2bfbits(val);
                    else
                        ((float*)Cp)[(size_t)m * ldc + n] = val;
                }
            }
}

// ---------------------------------------------------------------------------
// LayerNorm over compressed_kv (first 512 of 576 cols). One block per row.
// ---------------------------------------------------------------------------
__global__ __launch_bounds__(256) void ln_kernel(const float* __restrict__ ckv,
                                                 const float* __restrict__ g,
                                                 const float* __restrict__ b,
                                                 float* __restrict__ out) {
    const int row = blockIdx.x, tid = threadIdx.x;
    float2 x = *(const float2*)(ckv + (size_t)row * CKV_N + tid * 2);
    float s = x.x + x.y, s2 = x.x * x.x + x.y * x.y;
    #pragma unroll
    for (int off = 16; off; off >>= 1) {
        s  += __shfl_xor(s,  off, 32);
        s2 += __shfl_xor(s2, off, 32);
    }
    __shared__ float w1[8], w2[8];
    if ((tid & 31) == 0) { w1[tid >> 5] = s; w2[tid >> 5] = s2; }
    __syncthreads();
    float tot = 0.f, tot2 = 0.f;
    #pragma unroll
    for (int i = 0; i < 8; ++i) { tot += w1[i]; tot2 += w2[i]; }
    float mu  = tot * (1.f / KVLORA);
    float var = tot2 * (1.f / KVLORA) - mu * mu;
    float rs  = rsqrtf(var + 1e-5f);
    int c = tid * 2;
    out[(size_t)row * KVLORA + c]     = (x.x - mu) * rs * g[c]     + b[c];
    out[(size_t)row * KVLORA + c + 1] = (x.y - mu) * rs * g[c + 1] + b[c + 1];
}

// ---------------------------------------------------------------------------
// RoPE on q (reference interleave permutation), f32 q -> bf16 query.
// ---------------------------------------------------------------------------
__global__ __launch_bounds__(256) void rope_q_kernel(const float* __restrict__ q,
                                                     const float* __restrict__ cosc,
                                                     const float* __restrict__ sinc,
                                                     uint16_t* __restrict__ query) {
    int idx = blockIdx.x * 256 + threadIdx.x;      // over S*HQ
    int s = idx / HQ, col = idx % HQ, dd = col % QHEAD;
    float out;
    if (dd < DNOPE) {
        out = q[idx];
    } else {
        int j = dd - DNOPE;
        int base = idx - dd + DNOPE;               // s*HQ + h*QHEAD + 128
        float xi, rot;
        if (j < 32) { xi = q[base + 2 * j];             rot = -q[base + 2 * j + 1]; }
        else        { xi = q[base + 2 * (j - 32) + 1];  rot =  q[base + 2 * (j - 32)]; }
        out = xi * cosc[s * DROPE + j] + rot * sinc[s * DROPE + j];
    }
    query[idx] = f2bfbits(out);
}

// RoPE on k_pe (cols 512..575 of ckv), one rotated row per position.
__global__ __launch_bounds__(256) void rope_k_kernel(const float* __restrict__ ckv,
                                                     const float* __restrict__ cosc,
                                                     const float* __restrict__ sinc,
                                                     uint16_t* __restrict__ kpe) {
    int idx = blockIdx.x * 256 + threadIdx.x;      // over S*64
    int s = idx >> 6, j = idx & 63;
    size_t base = (size_t)s * CKV_N + KVLORA;
    float xi, rot;
    if (j < 32) { xi = ckv[base + 2 * j];            rot = -ckv[base + 2 * j + 1]; }
    else        { xi = ckv[base + 2 * (j - 32) + 1]; rot =  ckv[base + 2 * (j - 32)]; }
    kpe[idx] = f2bfbits(xi * cosc[s * DROPE + j] + rot * sinc[s * DROPE + j]);
}

// ---------------------------------------------------------------------------
// Flash attention. grid = (head, S/128). 8 waves, each owns 16 q rows.
// K processed in blocks of 32 keys: 12 WMMAs scores + 8 WMMAs P*V.
// ---------------------------------------------------------------------------
__global__ __launch_bounds__(256) void attn_kernel(const uint16_t* __restrict__ query,
                                                   const uint16_t* __restrict__ kv,
                                                   const uint16_t* __restrict__ kpe,
                                                   uint16_t* __restrict__ ctx) {
    __shared__ uint16_t Kt[32][200];    // 32 keys x 192 dims (+pad), 400B rows
    __shared__ uint16_t Vt[128][40];    // transposed V: 128 dims x 32 keys (+pad)
    __shared__ uint16_t Pt[8][16][32];  // per-wave P transpose buffer

    const int tid  = threadIdx.x;
    const int lane = tid & 31, wave = tid >> 5;
    const int hf   = lane >> 4, l16 = lane & 15;
    const int h = blockIdx.x, qb = blockIdx.y;
    const int qrow = qb * 128 + wave * 16 + l16;   // A-fragment row (m = l16)

    // Q fragments: 16 x 192 bf16 = 6 A-fragments.
    AFrag qf[6];
    #pragma unroll
    for (int t = 0; t < 6; ++t) {
        const uint16_t* p = query + (size_t)qrow * HQ + h * QHEAD + t * 32;
        qf[t].q[0] = *(const uint4*)(p + 8 * hf);
        qf[t].q[1] = *(const uint4*)(p + 16 + 8 * hf);
    }

    const v8f vz = {0.f, 0.f, 0.f, 0.f, 0.f, 0.f, 0.f, 0.f};
    float mrow[8], lrow[8];
    v8f O[8];
    #pragma unroll
    for (int r = 0; r < 8; ++r) { mrow[r] = -1e30f; lrow[r] = 0.f; }
    #pragma unroll
    for (int g = 0; g < 8; ++g) O[g] = vz;

    const int nkb = (qb + 1) * 4;                  // causal: keys <= qb*128+127
    for (int kb = 0; kb < nkb; ++kb) {
        const int kbase = kb * 32;

        // best-effort prefetch of next key block (K rows + V rows + rope)
        if (kb + 1 < nkb) {
            const int nb = (kb + 1) * 32;
            if (tid < 32) {
                const uint16_t* p = kv + (size_t)(nb + tid) * KV_N + h * 256;
                __builtin_prefetch(p, 0, 1);
                __builtin_prefetch(p + DNOPE, 0, 1);
            } else if (tid < 64) {
                __builtin_prefetch(kpe + (size_t)(nb + tid - 32) * DROPE, 0, 1);
            }
        }

        // stage K rows (nope from kv, rope from kpe): 384 x 16B chunks
        for (int c = tid; c < 384; c += 256) {
            int row = c / 12, col = c % 12;
            uint4 v;
            if (col < 8)
                v = *(const uint4*)(kv + (size_t)(kbase + row) * KV_N + h * 256 + col * 16);
            else
                v = *(const uint4*)(kpe + (size_t)(kbase + row) * DROPE + (col - 8) * 16);
            *(uint4*)&Kt[row][col * 16] = v;
        }
        // stage V transposed: 512 x 8-elem chunks, scalar scatter into Vt
        for (int c = tid; c < 512; c += 256) {
            int row = c >> 4, col8 = c & 15;
            uint4 v = *(const uint4*)(kv + (size_t)(kbase + row) * KV_N + h * 256 + DNOPE + col8 * 8);
            const uint16_t* e = (const uint16_t*)&v;
            #pragma unroll
            for (int i = 0; i < 8; ++i) Vt[col8 * 8 + i][row] = e[i];
        }
        __syncthreads();

        // scores: two 16x16 tiles over keys [kbase, kbase+32)
        v8f s0 = vz, s1 = vz;
        #pragma unroll
        for (int t = 0; t < 6; ++t) {
            const int d0 = t * 32;
            AFrag b0, b1;
            b0.q[0] = *(const uint4*)&Kt[l16][d0 + 8 * hf];
            b0.q[1] = *(const uint4*)&Kt[l16][d0 + 16 + 8 * hf];
            b1.q[0] = *(const uint4*)&Kt[16 + l16][d0 + 8 * hf];
            b1.q[1] = *(const uint4*)&Kt[16 + l16][d0 + 16 + 8 * hf];
            s0 = wmma_bf16(qf[t], b0, s0);
            s1 = wmma_bf16(qf[t], b1, s1);
        }

        // online softmax per row (row = r + 8*hf spread over 16 lanes)
        #pragma unroll
        for (int r = 0; r < 8; ++r) {
            const int sq = qb * 128 + wave * 16 + r + 8 * hf;
            float x0 = s0[r] * ATT_SCALE + ((kbase + l16)      > sq ? -1e9f : 0.f);
            float x1 = s1[r] * ATT_SCALE + ((kbase + 16 + l16) > sq ? -1e9f : 0.f);
            float mx = fmaxf(x0, x1);
            mx = fmaxf(mx, __shfl_xor(mx, 1, 32));
            mx = fmaxf(mx, __shfl_xor(mx, 2, 32));
            mx = fmaxf(mx, __shfl_xor(mx, 4, 32));
            mx = fmaxf(mx, __shfl_xor(mx, 8, 32));
            float mnew = fmaxf(mrow[r], mx);
            float p0 = __expf(x0 - mnew), p1 = __expf(x1 - mnew);
            float ps = p0 + p1;
            ps += __shfl_xor(ps, 1, 32);
            ps += __shfl_xor(ps, 2, 32);
            ps += __shfl_xor(ps, 4, 32);
            ps += __shfl_xor(ps, 8, 32);
            float alpha = __expf(mrow[r] - mnew);
            lrow[r] = lrow[r] * alpha + ps;
            mrow[r] = mnew;
            #pragma unroll
            for (int g = 0; g < 8; ++g) O[g][r] *= alpha;
            // C-layout -> LDS (row-major P) for A-layout reload
            Pt[wave][r + 8 * hf][l16]      = f2bfbits(p0);
            Pt[wave][r + 8 * hf][16 + l16] = f2bfbits(p1);
        }

        // reload P as 16x32 A fragment (same-wave LDS, in-order)
        AFrag pf;
        pf.q[0] = *(const uint4*)&Pt[wave][l16][8 * hf];
        pf.q[1] = *(const uint4*)&Pt[wave][l16][16 + 8 * hf];

        // O += P * V
        #pragma unroll
        for (int g = 0; g < 8; ++g) {
            AFrag bv;
            bv.q[0] = *(const uint4*)&Vt[g * 16 + l16][8 * hf];
            bv.q[1] = *(const uint4*)&Vt[g * 16 + l16][16 + 8 * hf];
            O[g] = wmma_bf16(pf, bv, O[g]);
        }
        __syncthreads();
    }

    // normalize + store context bf16: ctx[s][h*128 + d]
    #pragma unroll
    for (int g = 0; g < 8; ++g)
        #pragma unroll
        for (int r = 0; r < 8; ++r) {
            int sq = qb * 128 + wave * 16 + r + 8 * hf;
            ctx[(size_t)sq * CTX_N + h * DV + g * 16 + l16] = f2bfbits(O[g][r] / lrow[r]);
        }
}

// ---------------------------------------------------------------------------
// Host orchestration.
// ---------------------------------------------------------------------------
static inline size_t align256(size_t x) { return (x + 255) & ~(size_t)255; }

extern "C" void kernel_launch(void* const* d_in, const int* in_sizes, int n_in,
                              void* d_out, int out_size, void* d_ws, size_t ws_size,
                              hipStream_t stream) {
    (void)in_sizes; (void)n_in; (void)out_size; (void)ws_size;
    const float* hidden = (const float*)d_in[0];
    // d_in[1] attention_mask (causal, applied analytically), d_in[2] position_ids (= arange)
    const float* q_w    = (const float*)d_in[3];
    const float* kv_a_w = (const float*)d_in[4];
    const float* ln_g   = (const float*)d_in[5];
    const float* ln_b   = (const float*)d_in[6];
    const float* kv_b_w = (const float*)d_in[7];
    const float* o_w    = (const float*)d_in[8];
    const float* cosc   = (const float*)d_in[9];
    const float* sinc   = (const float*)d_in[10];
    float* out = (float*)d_out;

    char* ws = (char*)d_ws;
    size_t off = 0;
    float*    q_f32   = (float*)(ws + off);   off = align256(off + (size_t)S_LEN * HQ * 4);
    float*    ckv_f32 = (float*)(ws + off);   off = align256(off + (size_t)S_LEN * CKV_N * 4);
    float*    ln_f32  = (float*)(ws + off);   off = align256(off + (size_t)S_LEN * KVLORA * 4);
    uint16_t* kv_bf   = (uint16_t*)(ws + off); off = align256(off + (size_t)S_LEN * KV_N * 2);
    uint16_t* query   = (uint16_t*)(ws + off); off = align256(off + (size_t)S_LEN * HQ * 2);
    uint16_t* kpe     = (uint16_t*)(ws + off); off = align256(off + (size_t)S_LEN * DROPE * 2);
    uint16_t* ctx     = (uint16_t*)(ws + off); off = align256(off + (size_t)S_LEN * CTX_N * 2);

    dim3 blk(256);

    // 1) q = hidden @ q_w.T   (2048 x 6144 x 4096), f32 out
    gemm_nt<float, float, false><<<dim3(HQ / 128, S_LEN / 128), blk, 0, stream>>>(
        hidden, q_w, q_f32, S_LEN, HQ, HIDDEN, HIDDEN, HIDDEN, HQ);

    // 2) ckv = hidden @ kv_a_w.T (2048 x 576 x 4096), f32 out
    gemm_nt<float, float, false><<<dim3((CKV_N + 127) / 128, S_LEN / 128), blk, 0, stream>>>(
        hidden, kv_a_w, ckv_f32, S_LEN, CKV_N, HIDDEN, HIDDEN, HIDDEN, CKV_N);

    // 3) LayerNorm over compressed_kv
    ln_kernel<<<S_LEN, blk, 0, stream>>>(ckv_f32, ln_g, ln_b, ln_f32);

    // 4) RoPE on q -> bf16 query ; RoPE on k_pe -> bf16 kpe
    rope_q_kernel<<<(S_LEN * HQ) / 256, blk, 0, stream>>>(q_f32, cosc, sinc, query);
    rope_k_kernel<<<(S_LEN * DROPE) / 256, blk, 0, stream>>>(ckv_f32, cosc, sinc, kpe);

    // 5) kv = ln @ kv_b_w.T (2048 x 8192 x 512), bf16 out
    gemm_nt<float, float, true><<<dim3(KV_N / 128, S_LEN / 128), blk, 0, stream>>>(
        ln_f32, kv_b_w, kv_bf, S_LEN, KV_N, KVLORA, KVLORA, KVLORA, KV_N);

    // 6) flash attention -> bf16 context
    attn_kernel<<<dim3(NHEAD, S_LEN / 128), blk, 0, stream>>>(query, kv_bf, kpe, ctx);

    // 7) out = ctx @ o_w.T (2048 x 4096 x 4096), f32 out
    gemm_nt<bf16_t, float, false><<<dim3(HIDDEN / 128, S_LEN / 128), blk, 0, stream>>>(
        ctx, o_w, out, S_LEN, HIDDEN, CTX_N, CTX_N, CTX_N, HIDDEN);
}
